// ConferenceVLAGRAgent_35356170781175
// MI455X (gfx1250) — matrix operand order repaired and checked by reading, not compiled, not run on hardware
//
#include <hip/hip_runtime.h>
#include <stdint.h>

// ConferenceVLAGRAgent planner for MI455X (gfx1250, wave32, WMMA).
// bf16 16x16x32 WMMA with fp32 accumulation for all K=1024 GEMMs; weights
// converted once to bf16 in workspace (L2-resident thereafter). Sequential
// scans (ODE, GRU) are per-step kernels ordered by the stream.
// GEMM blocks are 4 waves sharing one A-slab staged in LDS via the gfx1250
// async global->LDS path (ASYNCcnt) when available.

#define TT 128
#define BB 64
#define HH 1024
#define NC 10

typedef __bf16 bf16_t;
typedef __attribute__((ext_vector_type(16))) __bf16 v16bf;
typedef __attribute__((ext_vector_type(8)))  __bf16 v8bf;
typedef __attribute__((ext_vector_type(8)))  float  v8f;

__device__ inline bf16_t f2bf(float x) {
  uint32_t u = __builtin_bit_cast(uint32_t, x);
  uint32_t r = u + 0x7FFFu + ((u >> 16) & 1u);   // round-to-nearest-even
  uint16_t h = (uint16_t)(r >> 16);
  return __builtin_bit_cast(bf16_t, h);
}
__device__ inline float bf2f(bf16_t h) {
  uint16_t u = __builtin_bit_cast(uint16_t, h);
  return __builtin_bit_cast(float, (uint32_t)u << 16);
}
__device__ inline float sigmoid_(float x) { return 1.0f / (1.0f + expf(-x)); }
__device__ inline float gelu_(float x) { return 0.5f * x * (1.0f + erff(x * 0.70710678118654752f)); }
__device__ inline float softplus_(float x) { return (x > 20.0f) ? x : log1pf(expf(x)); }

// ---- async global->LDS staging (GLOBAL_LOAD_ASYNC_TO_LDS_B128, ASYNCcnt) ----
#if __has_builtin(__builtin_amdgcn_global_load_async_to_lds_b128) && \
    __has_builtin(__builtin_amdgcn_s_wait_asynccnt)
typedef __attribute__((__vector_size__(4 * sizeof(int)))) int v4i_t;
typedef __attribute__((address_space(1))) v4i_t as1_v4i;
typedef __attribute__((address_space(3))) v4i_t as3_v4i;
__device__ inline void async_cp16(const void* g, void* l) {
  __builtin_amdgcn_global_load_async_to_lds_b128(
      (as1_v4i*)(v4i_t*)(char*)g, (as3_v4i*)(v4i_t*)(char*)l, 0, 0);
}
__device__ inline void async_cp_wait() { __builtin_amdgcn_s_wait_asynccnt(0); }
#else
__device__ inline void async_cp16(const void* g, void* l) {
  *(uint4*)l = *(const uint4*)g;           // fallback: b128 load + ds_store_b128
}
__device__ inline void async_cp_wait() {}
#endif

// Copy a 16-row x 1024-col bf16 slab (32KB) into LDS with all 128 threads.
__device__ inline void stage_slab(const bf16_t* __restrict__ src, bf16_t* __restrict__ dst,
                                  int tid) {
  const char* g = (const char*)src;
  char* l = (char*)dst;
#pragma unroll
  for (int i = 0; i < 16; ++i) {
    int off = (tid + i * 128) * 16;
    async_cp16(g + off, l + off);
  }
}

// ---- WMMA fragment loaders (contiguous k order, same for A and B) ----
// A (16-bit, 16x32): lane l holds row M=l%16; lanes<16: K0..15,
// lanes>=16: K16..31.
__device__ inline v16bf load_a_tile(const bf16_t* __restrict__ A, int lda,
                                    int m0, int k0, int lane) {
  const int row = m0 + (lane & 15);
  const int hi  = (lane >> 4) & 1;
  const bf16_t* p = A + (size_t)row * lda + k0 + hi * 16;
  const v8bf* pv = (const v8bf*)p;
  v8bf lo = pv[0];
  v8bf hi8 = pv[1];       // +8 elements
  v16bf a;
#pragma unroll
  for (int i = 0; i < 8; ++i) { a[i] = lo[i]; a[8 + i] = hi8[i]; }
  return a;
}
// B (16-bit, 32x16): lane l holds column N=l%16; lanes<16: K0..15,
// lanes>=16: K16..31.  W stored [N,K] row-major (torch Linear weight) so
// B = W^T loads are contiguous along K.
__device__ inline v16bf load_b_tile(const bf16_t* __restrict__ W, int ldw,
                                    int n0, int k0, int lane) {
  const int col = n0 + (lane & 15);
  const int hi  = (lane >> 4) & 1;
  const bf16_t* p = W + (size_t)col * ldw + k0 + hi * 16;
  const v8bf* pv = (const v8bf*)p;
  v8bf b0 = pv[0], b1 = pv[1];
  v16bf b;
#pragma unroll
  for (int i = 0; i < 8; ++i) { b[i] = b0[i]; b[8 + i] = b1[i]; }
  return b;
}
__device__ inline v8f wmma_bf16(v16bf a, v16bf b, v8f c) {
  return __builtin_amdgcn_wmma_f32_16x16x32_bf16(false, a, false, b, (short)0, c, false, false);
}

// ---------------- small utility kernels ----------------
__global__ void k_f2bf(const float* __restrict__ s, bf16_t* __restrict__ d, int n) {
  int i = blockIdx.x * 256 + threadIdx.x;
  if (i < n) d[i] = f2bf(s[i]);
}
__global__ void k_zerof(float* __restrict__ p, int n) {
  int i = blockIdx.x * 256 + threadIdx.x;
  if (i < n) p[i] = 0.0f;
}

// goal_pos = goal_emb @ goal_W.T + goal_b ; also seed geodesic[0] = start
__global__ void k_goal(const float* __restrict__ emb, const float* __restrict__ gW,
                       const float* __restrict__ gb, const float* __restrict__ start,
                       float* __restrict__ goal_pos, float* __restrict__ geodesic) {
  int tid = threadIdx.x;
  if (tid < 192) {
    int b = tid / 3, c = tid % 3;
    float acc = gb[c];
    const float* e = emb + (size_t)b * HH;
    const float* w = gW + (size_t)c * HH;
    for (int h = 0; h < HH; ++h) acc += e[h] * w[h];
    goal_pos[tid] = acc;
    geodesic[tid] = start[tid];
  }
}

// initial_path -> d_out[98304 ...]
__global__ void k_initpath(const float* __restrict__ start, const float* __restrict__ goal_pos,
                           float* __restrict__ dout) {
  int gid = blockIdx.x * 256 + threadIdx.x;
  if (gid >= BB * TT * 3) return;
  int b = gid / (TT * 3);
  int r = gid % (TT * 3);
  int t = r / 3, c = r % 3;
  float tv = (float)t * (1.0f / 127.0f);
  dout[98304 + gid] = start[b * 3 + c] * (1.0f - tv) + goal_pos[b * 3 + c] * tv;
}

// aff[b][c] = mean over 128x128 of uncertainty
__global__ void k_aff(const float* __restrict__ unc, float* __restrict__ aff) {
  int b = blockIdx.x, tid = threadIdx.x;
  const float* p = unc + (size_t)b * 16384 * 3;
  float a0 = 0, a1 = 0, a2 = 0;
  for (int i = tid; i < 16384; i += 256) {
    a0 += p[i * 3 + 0]; a1 += p[i * 3 + 1]; a2 += p[i * 3 + 2];
  }
  __shared__ float sm[768];
  sm[tid] = a0; sm[256 + tid] = a1; sm[512 + tid] = a2;
  __syncthreads();
  for (int s = 128; s > 0; s >>= 1) {
    if (tid < s) {
      sm[tid] += sm[tid + s];
      sm[256 + tid] += sm[256 + tid + s];
      sm[512 + tid] += sm[512 + tid + s];
    }
    __syncthreads();
  }
  if (tid == 0) {
    aff[b * 3 + 0] = sm[0]   * (1.0f / 16384.0f);
    aff[b * 3 + 1] = sm[256] * (1.0f / 16384.0f);
    aff[b * 3 + 2] = sm[512] * (1.0f / 16384.0f);
  }
}

// ODE step, phase 1: (t>0) finish p_t = p_{t-1} + dt*(h2 @ W3.T + b3);
// then h1 = gelu([p_t, noise_t] @ W1.T + b1) as bf16.  One block per batch b.
__global__ void k_ode_pre(int t, int do_h1,
                          float* __restrict__ geodesic, const float* __restrict__ noise,
                          const float* __restrict__ h2f,
                          const float* __restrict__ W1, const float* __restrict__ b1,
                          const float* __restrict__ W3, const float* __restrict__ b3,
                          bf16_t* __restrict__ h1bf) {
  const int b = blockIdx.x;
  const int tid = threadIdx.x;
  __shared__ float sv[16];
  __shared__ float redc[3];
  const float DT = 1.0f / 127.0f;
  if (t > 0) {
    if (tid < 96) {
      int c = tid >> 5, l = tid & 31;
      const float* hr = h2f + (size_t)b * HH;
      const float* wr = W3 + (size_t)c * HH;
      float acc = 0.0f;
      for (int k = l; k < HH; k += 32) acc += hr[k] * wr[k];
#pragma unroll
      for (int o = 16; o > 0; o >>= 1) acc += __shfl_down(acc, o);
      if (l == 0) redc[c] = acc;
    }
    __syncthreads();
    if (tid < 3) {
      float p = geodesic[(t - 1) * 192 + b * 3 + tid] + DT * (redc[tid] + b3[tid]);
      geodesic[t * 192 + b * 3 + tid] = p;
      sv[tid] = p;
    }
  } else {
    if (tid < 3) sv[tid] = geodesic[b * 3 + tid];
  }
  if (do_h1 && tid >= 3 && tid < 13) sv[tid] = noise[((size_t)t * BB + b) * NC + (tid - 3)];
  if (tid >= 13 && tid < 16) sv[tid] = 0.0f;
  __syncthreads();
  if (do_h1) {
    for (int j = tid; j < HH; j += 128) {
      const float* w = W1 + (size_t)j * 13;
      float acc = b1[j];
#pragma unroll
      for (int k = 0; k < 13; ++k) acc += sv[k] * w[k];
      h1bf[(size_t)b * HH + j] = f2bf(gelu_(acc));
    }
  }
}

// ODE step, phase 2 (WMMA): h2 = gelu(h1 @ W2.T + b2)  [64,1024]x[1024,1024].
// 128-thread blocks: 4 waves share one LDS-staged A slab; wave w -> j-tile.
__global__ void k_ode_mm(const bf16_t* __restrict__ h1bf, const bf16_t* __restrict__ W2bf,
                         const float* __restrict__ b2, float* __restrict__ h2f) {
  __shared__ bf16_t As[16 * HH];
  const int tid = threadIdx.x;
  const int lane = tid & 31, wave = tid >> 5;
  const int mt = blockIdx.x & 3, jg = blockIdx.x >> 2;    // grid = 64
  const int m0 = mt * 16;
  const int j0 = (jg * 4 + wave) * 16;
  stage_slab(h1bf + (size_t)m0 * HH, As, tid);
  async_cp_wait();
  __syncthreads();
  v8f acc = {};
  for (int k0 = 0; k0 < HH; k0 += 32) {
    v16bf a = load_a_tile(As, HH, 0, k0, lane);
    v16bf b = load_b_tile(W2bf, HH, j0, k0, lane);
    acc = wmma_bf16(a, b, acc);
  }
  const int hi = lane >> 4;
  const int jc = j0 + (lane & 15);
  const float bias = b2[jc];
#pragma unroll
  for (int v = 0; v < 8; ++v) {
    int row = m0 + v + 8 * hi;
    h2f[(size_t)row * HH + jc] = gelu_(acc[v] + bias);
  }
}

// Bilinear-sample metric at geodesic, build padded GRU input x0p[t][b][16].
__global__ void k_sample(const float* __restrict__ metric, const float* __restrict__ geodesic,
                         float* __restrict__ x0p) {
  int gid = blockIdx.x * 256 + threadIdx.x;
  if (gid >= TT * BB) return;
  int t = gid / BB, b = gid % BB;
  const float* g = geodesic + t * 192 + b * 3;
  float fx = ((g[0] * 0.1f + 1.0f) * 128.0f - 1.0f) * 0.5f;
  float fy = ((g[2] * 0.1f + 1.0f) * 128.0f - 1.0f) * 0.5f;
  float x0 = floorf(fx), y0 = floorf(fy);
  float wx1 = fx - x0, wy1 = fy - y0;
  int xi0 = (int)x0, yi0 = (int)y0;
  float outc[NC];
#pragma unroll
  for (int c = 0; c < NC; ++c) outc[c] = 0.0f;
#pragma unroll
  for (int dx = 0; dx < 2; ++dx) {
    float wx = dx ? wx1 : 1.0f - wx1;
    int xi = xi0 + dx;
#pragma unroll
    for (int dy = 0; dy < 2; ++dy) {
      float wy = dy ? wy1 : 1.0f - wy1;
      int yi = yi0 + dy;
      bool valid = (xi >= 0) && (xi < 128) && (yi >= 0) && (yi < 128);
      int xc = xi < 0 ? 0 : (xi > 127 ? 127 : xi);
      int yc = yi < 0 ? 0 : (yi > 127 ? 127 : yi);
      const float* m = metric + (((size_t)b * 128 + yc) * 128 + xc) * NC;
      float w = wx * wy * (valid ? 1.0f : 0.0f);
#pragma unroll
      for (int c = 0; c < NC; ++c) outc[c] += m[c] * w;
    }
  }
  float* o = x0p + (size_t)gid * 16;
  o[0] = g[0]; o[1] = g[1]; o[2] = g[2];
#pragma unroll
  for (int c = 0; c < NC; ++c) o[3 + c] = outc[c];
  o[13] = 0.0f; o[14] = 0.0f; o[15] = 0.0f;
}

// One GRU timestep, gates fused.  grid = 64 blocks x 128 threads (4 waves).
// Block = one m-tile x 4 j-tiles; the h_prev (and layer1: x) A-slabs are
// staged once into LDS (async copy), then each wave computes 16x16 tiles of
// the r/z/n gate columns (gh via WMMA over K=1024; gi via WMMA for layer1 or
// scalar K=13 for layer0).
__global__ void k_gru_step(const bf16_t* __restrict__ hprev_bf,
                           const float*  __restrict__ hprev_f,
                           const bf16_t* __restrict__ Whh_bf,
                           const bf16_t* __restrict__ Wih_bf,
                           const bf16_t* __restrict__ x_bf,
                           const float*  __restrict__ x0p_t,
                           const float*  __restrict__ Wih0,
                           const float*  __restrict__ bih,
                           const float*  __restrict__ bhh,
                           float* __restrict__ hout_f,
                           bf16_t* __restrict__ hout_bf,
                           float* __restrict__ out_c3,
                           int layer) {
  __shared__ bf16_t As[16 * HH];
  __shared__ bf16_t Xs[16 * HH];
  const int tid = threadIdx.x;
  const int lane = tid & 31, wave = tid >> 5;
  const int mt = blockIdx.x & 3;
  const int jg = blockIdx.x >> 2;                 // grid = 64
  const int m0 = mt * 16;
  const int j0 = (jg * 4 + wave) * 16;
  const int hi = lane >> 4;
  const int cl = lane & 15;

  stage_slab(hprev_bf + (size_t)m0 * HH, As, tid);
  if (layer == 1) stage_slab(x_bf + (size_t)m0 * HH, Xs, tid);
  async_cp_wait();
  __syncthreads();

  v8f gh[3], gi[3];
#pragma unroll
  for (int g = 0; g < 3; ++g) { gh[g] = {}; gi[g] = {}; }

  for (int k0 = 0; k0 < HH; k0 += 32) {
    v16bf a = load_a_tile(As, HH, 0, k0, lane);
#pragma unroll
    for (int g = 0; g < 3; ++g) {
      v16bf b = load_b_tile(Whh_bf, HH, g * HH + j0, k0, lane);
      gh[g] = wmma_bf16(a, b, gh[g]);
    }
  }
  if (layer == 1) {
    for (int k0 = 0; k0 < HH; k0 += 32) {
      v16bf a = load_a_tile(Xs, HH, 0, k0, lane);
#pragma unroll
      for (int g = 0; g < 3; ++g) {
        v16bf b = load_b_tile(Wih_bf, HH, g * HH + j0, k0, lane);
        gi[g] = wmma_bf16(a, b, gi[g]);
      }
    }
  } else {
#pragma unroll
    for (int g = 0; g < 3; ++g) {
      const float* wr = Wih0 + (size_t)(g * HH + j0 + cl) * 13;
      float w[13];
#pragma unroll
      for (int k = 0; k < 13; ++k) w[k] = wr[k];
#pragma unroll
      for (int v = 0; v < 8; ++v) {
        int row = m0 + v + 8 * hi;
        const float* xr = x0p_t + (size_t)row * 16;
        float acc = 0.0f;
#pragma unroll
        for (int k = 0; k < 13; ++k) acc += xr[k] * w[k];
        gi[g][v] = acc;
      }
    }
  }
  const int jc = j0 + cl;
  const float bi_r = bih[jc],          bh_r = bhh[jc];
  const float bi_z = bih[HH + jc],     bh_z = bhh[HH + jc];
  const float bi_n = bih[2 * HH + jc], bh_n = bhh[2 * HH + jc];
#pragma unroll
  for (int v = 0; v < 8; ++v) {
    int row = m0 + v + 8 * hi;
    float r  = sigmoid_(gi[0][v] + bi_r + gh[0][v] + bh_r);
    float z  = sigmoid_(gi[1][v] + bi_z + gh[1][v] + bh_z);
    float nn = tanhf(gi[2][v] + bi_n + r * (gh[2][v] + bh_n));
    float hp = hprev_f[(size_t)row * HH + jc];
    float h2 = (1.0f - z) * nn + z * hp;
    hout_f[(size_t)row * HH + jc] = h2;
    hout_bf[(size_t)row * HH + jc] = f2bf(h2);
    if (out_c3 && jc < 3) out_c3[row * 3 + jc] = h2;
  }
}

// Uncertainty head GEMM1: relu(out1 @ unc_W1.T + b1), [8192,1024]x[1024,512].
// 4-wave blocks with LDS-staged A slab; grid = 512 m-tiles x 8 j-groups.
__global__ void k_unc1(const bf16_t* __restrict__ Xbf, const bf16_t* __restrict__ Wbf,
                       const float* __restrict__ b, bf16_t* __restrict__ Ybf) {
  __shared__ bf16_t As[16 * HH];
  const int tid = threadIdx.x;
  const int lane = tid & 31, wave = tid >> 5;
  const int mt = blockIdx.x & 511, jg = blockIdx.x >> 9;   // grid = 4096
  const int m0 = mt * 16;
  const int j0 = (jg * 4 + wave) * 16;
  stage_slab(Xbf + (size_t)m0 * HH, As, tid);
  async_cp_wait();
  __syncthreads();
  v8f acc = {};
  for (int k0 = 0; k0 < HH; k0 += 32) {
    v16bf a = load_a_tile(As, HH, 0, k0, lane);
    v16bf bm = load_b_tile(Wbf, HH, j0, k0, lane);
    acc = wmma_bf16(a, bm, acc);
  }
  const int hi = lane >> 4;
  const int jc = j0 + (lane & 15);
  const float bias = b[jc];
#pragma unroll
  for (int v = 0; v < 8; ++v) {
    int row = m0 + v + 8 * hi;
    float x = acc[v] + bias;
    Ybf[(size_t)row * 512 + jc] = f2bf(x > 0.0f ? x : 0.0f);
  }
}

// Uncertainty head GEMM2 (N=9, scalar): u = U1 @ unc_W2.T + b2
__global__ void k_unc2(const bf16_t* __restrict__ U1bf, const float* __restrict__ W2,
                       const float* __restrict__ b2, float* __restrict__ u) {
  int gid = blockIdx.x * 256 + threadIdx.x;
  if (gid >= TT * BB * 9) return;
  int bt = gid / 9, j = gid % 9;
  const bf16_t* x = U1bf + (size_t)bt * 512;
  const float* w = W2 + (size_t)j * 512;
  float acc = b2[j];
  for (int k = 0; k < 512; ++k) acc += bf2f(x[k]) * w[k];
  u[gid] = acc;
}

// refined, cov (L L^T scaled by 1+aff rows), diff
__global__ void k_final(const float* __restrict__ geodesic, const float* __restrict__ out1c3,
                        const float* __restrict__ u, const float* __restrict__ aff,
                        float* __restrict__ dout) {
  int gid = blockIdx.x * 256 + threadIdx.x;
  if (gid >= BB * TT) return;
  int b = gid / TT, t = gid % TT;
  const float* uu = u + ((size_t)(t * BB + b)) * 9;
  float a = softplus_(uu[0]), d = softplus_(uu[2]), f = softplus_(uu[5]);
  float l10 = uu[1], l20 = uu[3], l21 = uu[4];
  float cov[9];
  cov[0] = a * a;        cov[1] = a * l10;              cov[2] = a * l20;
  cov[3] = l10 * a;      cov[4] = l10 * l10 + d * d;    cov[5] = l10 * l20 + d * l21;
  cov[6] = l20 * a;      cov[7] = l20 * l10 + l21 * d;  cov[8] = l20 * l20 + l21 * l21 + f * f;
  float s0 = 1.0f + aff[b * 3 + 0], s1 = 1.0f + aff[b * 3 + 1], s2 = 1.0f + aff[b * 3 + 2];
  cov[0] *= s0; cov[1] *= s0; cov[2] *= s0;
  cov[3] *= s1; cov[4] *= s1; cov[5] *= s1;
  cov[6] *= s2; cov[7] *= s2; cov[8] *= s2;
  float* cv = dout + 24576 + (size_t)b * TT * 9 + (size_t)t * 9;
#pragma unroll
  for (int k = 0; k < 9; ++k) cv[k] = cov[k];
#pragma unroll
  for (int c = 0; c < 3; ++c) {
    float ref = geodesic[t * 192 + b * 3 + c] + 0.1f * out1c3[t * 192 + b * 3 + c];
    int pi = b * TT * 3 + t * 3 + c;
    dout[pi] = ref;
    dout[122880 + pi] = ref - dout[98304 + pi];
  }
}

extern "C" void kernel_launch(void* const* d_in, const int* in_sizes, int n_in,
                              void* d_out, int out_size, void* d_ws, size_t ws_size,
                              hipStream_t stream) {
  (void)in_sizes; (void)n_in; (void)out_size; (void)ws_size;
  const float* metric = (const float*)d_in[0];
  const float* start  = (const float*)d_in[1];
  const float* gemb   = (const float*)d_in[2];
  const float* unc_in = (const float*)d_in[4];
  const float* noise  = (const float*)d_in[5];
  const float* goal_W = (const float*)d_in[6];
  const float* goal_b = (const float*)d_in[7];
  const float* dyn_W1 = (const float*)d_in[8];
  const float* dyn_b1 = (const float*)d_in[9];
  const float* dyn_W2 = (const float*)d_in[10];
  const float* dyn_b2 = (const float*)d_in[11];
  const float* dyn_W3 = (const float*)d_in[12];
  const float* dyn_b3 = (const float*)d_in[13];
  const float* Wih0   = (const float*)d_in[14];
  const float* Whh0   = (const float*)d_in[15];
  const float* bih0   = (const float*)d_in[16];
  const float* bhh0   = (const float*)d_in[17];
  const float* Wih1   = (const float*)d_in[18];
  const float* Whh1   = (const float*)d_in[19];
  const float* bih1   = (const float*)d_in[20];
  const float* bhh1   = (const float*)d_in[21];
  const float* uW1    = (const float*)d_in[22];
  const float* ub1    = (const float*)d_in[23];
  const float* uW2    = (const float*)d_in[24];
  const float* ub2    = (const float*)d_in[25];
  float* out = (float*)d_out;

  char* ws = (char*)d_ws;
  size_t off = 0;
  auto take = [&](size_t bytes) -> char* {
    char* p = ws + off;
    off = (off + bytes + 255) & ~(size_t)255;
    return p;
  };
  bf16_t* W2bf   = (bf16_t*)take((size_t)1024 * 1024 * 2);
  bf16_t* Whh0bf = (bf16_t*)take((size_t)3072 * 1024 * 2);
  bf16_t* Wih1bf = (bf16_t*)take((size_t)3072 * 1024 * 2);
  bf16_t* Whh1bf = (bf16_t*)take((size_t)3072 * 1024 * 2);
  bf16_t* uW1bf  = (bf16_t*)take((size_t)512 * 1024 * 2);
  float*  goal_pos = (float*)take(192 * 4);
  float*  aff      = (float*)take(192 * 4);
  float*  geodesic = (float*)take((size_t)TT * BB * 3 * 4);
  bf16_t* h1bf     = (bf16_t*)take((size_t)BB * HH * 2);
  float*  h2f      = (float*)take((size_t)BB * HH * 4);
  float*  x0p      = (float*)take((size_t)TT * BB * 16 * 4);
  float*  hbuf0    = (float*)take((size_t)BB * HH * 4);
  float*  hbuf1    = (float*)take((size_t)BB * HH * 4);
  bf16_t* zbf      = (bf16_t*)take((size_t)BB * HH * 2);
  bf16_t* out0bf   = (bf16_t*)take((size_t)TT * BB * HH * 2);
  bf16_t* out1bf   = (bf16_t*)take((size_t)TT * BB * HH * 2);
  float*  out1c3   = (float*)take((size_t)TT * BB * 3 * 4);
  bf16_t* U1bf     = (bf16_t*)take((size_t)TT * BB * 512 * 2);
  float*  ubuf     = (float*)take((size_t)TT * BB * 9 * 4);

  // Weight conversions (one HBM pass; bf16 copies live in L2 thereafter).
  k_f2bf<<<(1024 * 1024 + 255) / 256, 256, 0, stream>>>(dyn_W2, W2bf, 1024 * 1024);
  k_f2bf<<<(3072 * 1024 + 255) / 256, 256, 0, stream>>>(Whh0, Whh0bf, 3072 * 1024);
  k_f2bf<<<(3072 * 1024 + 255) / 256, 256, 0, stream>>>(Wih1, Wih1bf, 3072 * 1024);
  k_f2bf<<<(3072 * 1024 + 255) / 256, 256, 0, stream>>>(Whh1, Whh1bf, 3072 * 1024);
  k_f2bf<<<(512 * 1024 + 255) / 256, 256, 0, stream>>>(uW1, uW1bf, 512 * 1024);
  k_zerof<<<256, 256, 0, stream>>>(hbuf0, BB * HH);
  k_zerof<<<128, 256, 0, stream>>>((float*)zbf, BB * HH / 2);

  k_goal<<<1, 256, 0, stream>>>(gemb, goal_W, goal_b, start, goal_pos, geodesic);
  k_initpath<<<96, 256, 0, stream>>>(start, goal_pos, out);
  k_aff<<<64, 256, 0, stream>>>(unc_in, aff);

  // Neural-ODE scan (sequential; 2 kernels per step).
  for (int t = 0; t < 127; ++t) {
    k_ode_pre<<<64, 128, 0, stream>>>(t, 1, geodesic, noise, h2f,
                                      dyn_W1, dyn_b1, dyn_W3, dyn_b3, h1bf);
    k_ode_mm<<<64, 128, 0, stream>>>(h1bf, W2bf, dyn_b2, h2f);
  }
  k_ode_pre<<<64, 128, 0, stream>>>(127, 0, geodesic, noise, h2f,
                                    dyn_W1, dyn_b1, dyn_W3, dyn_b3, h1bf);

  k_sample<<<(TT * BB + 255) / 256, 256, 0, stream>>>(metric, geodesic, x0p);

  // GRU layer 0 (sequential over t; fused gate kernel, WMMA gh).
  for (int t = 0; t < TT; ++t) {
    const bf16_t* hpbf = (t == 0) ? zbf : out0bf + (size_t)(t - 1) * BB * HH;
    const float* hpf = (t & 1) ? hbuf1 : hbuf0;
    float* hof = (t & 1) ? hbuf0 : hbuf1;
    k_gru_step<<<64, 128, 0, stream>>>(hpbf, hpf, Whh0bf,
                                       (const bf16_t*)nullptr, zbf,
                                       x0p + (size_t)t * BB * 16, Wih0, bih0, bhh0,
                                       hof, out0bf + (size_t)t * BB * HH,
                                       (float*)nullptr, 0);
  }
  // GRU layer 1 (WMMA for both gi and gh).
  k_zerof<<<256, 256, 0, stream>>>(hbuf0, BB * HH);
  for (int t = 0; t < TT; ++t) {
    const bf16_t* hpbf = (t == 0) ? zbf : out1bf + (size_t)(t - 1) * BB * HH;
    const float* hpf = (t & 1) ? hbuf1 : hbuf0;
    float* hof = (t & 1) ? hbuf0 : hbuf1;
    k_gru_step<<<64, 128, 0, stream>>>(hpbf, hpf, Whh1bf, Wih1bf,
                                       out0bf + (size_t)t * BB * HH,
                                       (const float*)nullptr, (const float*)nullptr,
                                       bih1, bhh1,
                                       hof, out1bf + (size_t)t * BB * HH,
                                       out1c3 + (size_t)t * BB * 3, 1);
  }

  // Uncertainty head + final assembly.
  k_unc1<<<4096, 128, 0, stream>>>(out1bf, uW1bf, ub1, U1bf);
  k_unc2<<<(TT * BB * 9 + 255) / 256, 256, 0, stream>>>(U1bf, uW2, ub2, ubuf);
  k_final<<<(TT * BB + 255) / 256, 256, 0, stream>>>(geodesic, out1c3, ubuf, aff, out);
}
